// Model_81724637709020
// MI455X (gfx1250) — matrix operand order repaired and checked by reading, not compiled, not run on hardware
//
#include <hip/hip_runtime.h>
#include <cstdint>
#include <cstddef>

// ---------------------------------------------------------------------------
// Problem dims (fixed by the reference)
// ---------------------------------------------------------------------------
#define B_   32
#define T_   400
#define S_   64
#define H_   512
#define V_   32000
#define L_   5
#define G4H  2048          // 4*H
#define TK   12800         // T_*B_  (flattened t,k contraction length)

// ---------------------------------------------------------------------------
// CDNA5 WMMA helpers (wave32, v_wmma_f32_16x16x32_bf16)
// ---------------------------------------------------------------------------
typedef __attribute__((ext_vector_type(16))) __bf16          v16bf;
typedef __attribute__((ext_vector_type(8)))  float           v8f;
typedef __attribute__((ext_vector_type(8)))  unsigned short  v8us;
typedef __attribute__((ext_vector_type(16))) unsigned short  v16us;
typedef unsigned short bfbits;   // bf16 stored as raw bits (host-portable)

__device__ __forceinline__ bfbits f2bf(float f) {
  unsigned u = __builtin_bit_cast(unsigned, f);
  u += 0x7FFFu + ((u >> 16) & 1u);            // round-to-nearest-even
  return (bfbits)(u >> 16);
}
__device__ __forceinline__ float sigmoidf_(float x) {
  return 1.0f / (1.0f + __expf(-x));
}
__device__ __forceinline__ v8f zero8() {
  v8f z;
#pragma unroll
  for (int i = 0; i < 8; ++i) z[i] = 0.0f;
  return z;
}

// A tile: 16x32 bf16, row-major source with leading dim `ld`.
// CDNA5 layout: lanes 0-15 -> row=lane, K={0..7,16..23};
//               lanes 16-31 -> row=lane-16, K={8..15,24..31}.
// Works for both global and LDS provenance (addrspace inferred after inlining).
__device__ __forceinline__ v16bf load_a_tile(const bfbits* a, int ld, int lane) {
  const int m  = lane & 15;
  const int k0 = (lane >> 4) << 3;            // 0 or 8
  const bfbits* p = a + (size_t)m * ld + k0;
  v8us lo = *(const v8us*)(p);                // K k0..k0+7
  v8us hi = *(const v8us*)(p + 16);           // K k0+16..k0+23
  v16us c = __builtin_shufflevector(lo, hi, 0,1,2,3,4,5,6,7,8,9,10,11,12,13,14,15);
  return __builtin_bit_cast(v16bf, c);
}

// B tile: 32x16 bf16 (KxN), stored transposed bt[n][k] (N rows x K cols, ld).
// CDNA5 layout: lanes 0-15 -> col=lane, K={0..7,16..23}; lanes 16-31 -> col=lane-16, K={8..15,24..31}.
__device__ __forceinline__ v16bf load_b_tile(const bfbits* bt, int ld, int lane) {
  const int n  = lane & 15;
  const int k0 = (lane >> 4) << 3;            // 0 or 8
  const bfbits* p = bt + (size_t)n * ld + k0;
  v8us lo = *(const v8us*)(p);
  v8us hi = *(const v8us*)(p + 16);
  v16us c = __builtin_shufflevector(lo, hi, 0,1,2,3,4,5,6,7,8,9,10,11,12,13,14,15);
  return __builtin_bit_cast(v16bf, c);
}

__device__ __forceinline__ v8f wmma_bf16(v16bf a, v16bf b, v8f c) {
  return __builtin_amdgcn_wmma_f32_16x16x32_bf16(false, a, false, b, (short)0, c,
                                                 false, false);
}

// ---------------------------------------------------------------------------
// fp32 -> bf16 bulk conversion
// ---------------------------------------------------------------------------
__global__ __launch_bounds__(256) void f2bf_kernel(const float* __restrict__ in,
                                                   bfbits* __restrict__ out,
                                                   size_t n) {
  size_t i = (size_t)blockIdx.x * 256 + threadIdx.x;
  if (i < n) out[i] = f2bf(in[i]);
}

// xs_bf[t][b][e] = bf16(emb[original[b][t]][e])
__global__ __launch_bounds__(256) void embed_all_kernel(const int* __restrict__ orig,
                                                        const float* __restrict__ emb,
                                                        bfbits* __restrict__ xs) {
  size_t i = (size_t)blockIdx.x * 256 + threadIdx.x;
  if (i >= (size_t)T_ * B_ * H_) return;
  int e = (int)(i & (H_ - 1));
  size_t tb = i >> 9;
  int b = (int)(tb & (B_ - 1));
  int t = (int)(tb >> 5);
  int tok = orig[(size_t)b * T_ + t];
  xs[i] = f2bf(emb[(size_t)tok * H_ + e]);
}

// Decoder-step embedding gather: x_bf[b][e] = bf16(emb[tok][e])
__global__ __launch_bounds__(256) void embed_step_kernel(const int* __restrict__ wtok,
                                                         const float* __restrict__ emb,
                                                         bfbits* __restrict__ x,
                                                         int use_bos) {
  int i = blockIdx.x * 256 + threadIdx.x;          // 0 .. B*H-1
  if (i >= B_ * H_) return;
  int e = i & (H_ - 1);
  int b = i >> 9;
  int tok = use_bos ? 1 : wtok[b];
  x[i] = f2bf(emb[(size_t)tok * H_ + e]);
}

// ---------------------------------------------------------------------------
// Persistent encoder: full T-step LSTM recurrence in ONE kernel.
// Grid = 2 blocks x 1024 threads (32 wave32 waves = one full WGP each):
//   block 0 -> forward chain, block 1 -> backward chain (concurrent WGPs).
// h lives in LDS (bf16, 32 KB) and is handed between steps with workgroup
// barriers; c lives entirely in VGPRs (the (m,n)->lane mapping is static, so
// each lane privately owns its 16 cell states). Weights stream from L2.
// Wave w owns 16 columns of H across all four gates -> lane-local nonlinearity.
// ---------------------------------------------------------------------------
__global__ __launch_bounds__(1024) void lstm_seq_kernel(
    const bfbits* __restrict__ xs,      // [T,B,H] bf16
    const bfbits* __restrict__ WihF, const bfbits* __restrict__ WhhF,
    const float*  __restrict__ biasF,
    const bfbits* __restrict__ WihB, const bfbits* __restrict__ WhhB,
    const float*  __restrict__ biasB,
    float* __restrict__ memF,           // [T,B,H] per-direction hidden states
    float* __restrict__ memB,
    float* __restrict__ hTF,            // [B,H] final hidden (fp32)
    float* __restrict__ hTB) {
  const int rev = blockIdx.x;
  const bfbits* Wih  = rev ? WihB : WihF;
  const bfbits* Whh  = rev ? WhhB : WhhF;
  const float*  bias = rev ? biasB : biasF;
  float* memO = rev ? memB : memF;
  float* hT   = rev ? hTB : hTF;

  __shared__ bfbits h_lds[B_ * H_];     // 32 KB of the 320 KB WGP LDS

  const int wave = threadIdx.x >> 5;    // 0..31
  const int lane = threadIdx.x & 31;
  const int col0 = wave * 16;           // 16 columns of H per wave
  const int n    = col0 + (lane & 15);
  const int mhi  = (lane >> 4) * 8;

  const float bi = bias[n];
  const float bf = bias[H_ + n];
  const float bg = bias[2 * H_ + n];
  const float bo = bias[3 * H_ + n];

  for (int i = threadIdx.x; i < B_ * H_; i += 1024) h_lds[i] = 0;
  __syncthreads();

  float creg[2][8];
#pragma unroll
  for (int j = 0; j < 2; ++j)
#pragma unroll
    for (int r = 0; r < 8; ++r) creg[j][r] = 0.0f;

#pragma unroll 1
  for (int step = 0; step < T_; ++step) {
    const int t = rev ? (T_ - 1 - step) : step;
    const bfbits* x = xs + (size_t)t * B_ * H_;
    if (step + 1 < T_) {
      const int tn = rev ? (T_ - 2 - step) : (step + 1);
      __builtin_prefetch(xs + (size_t)tn * B_ * H_ + (threadIdx.x << 4), 0, 1);
    }

    v8f acc[4][2];
#pragma unroll
    for (int q = 0; q < 4; ++q)
#pragma unroll
      for (int j = 0; j < 2; ++j) acc[q][j] = zero8();

    // ---- x @ Wih^T (A from global/L2) ----
#pragma unroll 1
    for (int kk = 0; kk < H_; kk += 32) {
      v16bf a0 = load_a_tile(x + kk, H_, lane);
      v16bf a1 = load_a_tile(x + (size_t)16 * H_ + kk, H_, lane);
#pragma unroll
      for (int q = 0; q < 4; ++q) {
        v16bf b = load_b_tile(Wih + (size_t)(q * H_ + col0) * H_ + kk, H_, lane);
        acc[q][0] = wmma_bf16(a0, b, acc[q][0]);
        acc[q][1] = wmma_bf16(a1, b, acc[q][1]);
      }
    }
    // ---- h @ Whh^T (A from LDS -> ds_load feeding WMMA) ----
#pragma unroll 1
    for (int kk = 0; kk < H_; kk += 32) {
      v16bf a0 = load_a_tile(h_lds + kk, H_, lane);
      v16bf a1 = load_a_tile(h_lds + 16 * H_ + kk, H_, lane);
#pragma unroll
      for (int q = 0; q < 4; ++q) {
        v16bf b = load_b_tile(Whh + (size_t)(q * H_ + col0) * H_ + kk, H_, lane);
        acc[q][0] = wmma_bf16(a0, b, acc[q][0]);
        acc[q][1] = wmma_bf16(a1, b, acc[q][1]);
      }
    }
    __syncthreads();   // all reads of h_lds complete before overwrite

#pragma unroll
    for (int j = 0; j < 2; ++j) {
#pragma unroll
      for (int r = 0; r < 8; ++r) {
        int m = j * 16 + mhi + r;
        float iv = sigmoidf_(acc[0][j][r] + bi);
        float fv = sigmoidf_(acc[1][j][r] + bf);
        float gv = tanhf   (acc[2][j][r] + bg);
        float ov = sigmoidf_(acc[3][j][r] + bo);
        float cn = fv * creg[j][r] + iv * gv;
        float hn = ov * tanhf(cn);
        creg[j][r] = cn;
        int idx = m * H_ + n;
        h_lds[idx] = f2bf(hn);
        memO[(size_t)t * B_ * H_ + idx] = hn;
        if (step == T_ - 1) hT[idx] = hn;
      }
    }
    __syncthreads();   // h_lds writes visible before next step's reads
  }
}

// ---------------------------------------------------------------------------
// Fused LSTM cell (single step, used by the decoder):
//   gates = x @ Wih^T + h @ Whh^T + b ; i,f,g,o nonlinearity ; c,h update.
// Grid: 4 blocks x 256 threads = 32 waves.
// ---------------------------------------------------------------------------
__global__ __launch_bounds__(256) void lstm_cell_kernel(
    const bfbits* __restrict__ x_bf,    // [B,H]
    const bfbits* __restrict__ h_bf,    // [B,H] (previous h)
    const bfbits* __restrict__ Wih,     // [4H,H] bf16
    const bfbits* __restrict__ Whh,     // [4H,H] bf16
    const float*  __restrict__ bias,    // [4H]
    float*        __restrict__ c,       // [B,H] in/out
    float*        __restrict__ h_out,   // [B,H]
    bfbits*       __restrict__ hbf_out) // [B,H]
{
  const int wave = (int)((blockIdx.x * blockDim.x + threadIdx.x) >> 5);  // 0..31
  const int lane = threadIdx.x & 31;
  const int col0 = wave * 16;

  v8f acc[4][2];
#pragma unroll
  for (int q = 0; q < 4; ++q)
#pragma unroll
    for (int j = 0; j < 2; ++j) acc[q][j] = zero8();

#pragma unroll 1
  for (int part = 0; part < 2; ++part) {
    const bfbits* A = part ? h_bf : x_bf;
    const bfbits* W = part ? Whh : Wih;
#pragma unroll 1
    for (int kk = 0; kk < H_; kk += 32) {
      v16bf a0 = load_a_tile(A + kk, H_, lane);
      v16bf a1 = load_a_tile(A + (size_t)16 * H_ + kk, H_, lane);
#pragma unroll
      for (int q = 0; q < 4; ++q) {
        v16bf b = load_b_tile(W + (size_t)(q * H_ + col0) * H_ + kk, H_, lane);
        acc[q][0] = wmma_bf16(a0, b, acc[q][0]);
        acc[q][1] = wmma_bf16(a1, b, acc[q][1]);
      }
    }
  }

  const int n   = col0 + (lane & 15);
  const int mhi = (lane >> 4) * 8;
#pragma unroll
  for (int j = 0; j < 2; ++j) {
#pragma unroll
    for (int r = 0; r < 8; ++r) {
      int m = j * 16 + mhi + r;
      float iv = sigmoidf_(acc[0][j][r] + bias[n]);
      float fv = sigmoidf_(acc[1][j][r] + bias[H_ + n]);
      float gv = tanhf   (acc[2][j][r] + bias[2 * H_ + n]);
      float ov = sigmoidf_(acc[3][j][r] + bias[3 * H_ + n]);
      size_t idx = (size_t)m * H_ + n;
      float cn = fv * c[idx] + iv * gv;
      float hn = ov * tanhf(cn);
      c[idx]       = cn;
      h_out[idx]   = hn;
      hbf_out[idx] = f2bf(hn);
    }
  }
}

// ---------------------------------------------------------------------------
// Generic M=32 WMMA GEMM:  out[32,N] = A[32,K](bf16) * B[K,N] (+bias)
// B provided transposed as Bt[N][K] row-major. One wave per 16x16 tile.
// ---------------------------------------------------------------------------
__global__ __launch_bounds__(256) void gemm_m32_kernel(
    const bfbits* __restrict__ A, const bfbits* __restrict__ Bt,
    const float* __restrict__ bias, float* __restrict__ outF,
    bfbits* __restrict__ outB, int N, int K) {
  const int w    = (int)((blockIdx.x * blockDim.x + threadIdx.x) >> 5);
  const int lane = threadIdx.x & 31;
  const int ntile = w >> 1;
  const int mtile = w & 1;
  if (ntile * 16 >= N) return;

  const bfbits* Arow = A + (size_t)mtile * 16 * K;
  const bfbits* Bcol = Bt + (size_t)(ntile * 16) * K;
  v8f acc = zero8();
#pragma unroll 1
  for (int kk = 0; kk < K; kk += 32) {
    if (kk + 2048 < K) __builtin_prefetch(Bcol + kk + 2048, 0, 1);
    v16bf a = load_a_tile(Arow + kk, K, lane);
    v16bf b = load_b_tile(Bcol + kk, K, lane);
    acc = wmma_bf16(a, b, acc);
  }

  const int n = ntile * 16 + (lane & 15);
  const int mb = mtile * 16 + (lane >> 4) * 8;
#pragma unroll
  for (int r = 0; r < 8; ++r) {
    int m = mb + r;
    float v = acc[r] + (bias ? bias[n] : 0.0f);
    if (outF) outF[(size_t)m * N + n] = v;
    if (outB) outB[(size_t)m * N + n] = f2bf(v);
  }
}

// ---------------------------------------------------------------------------
// beta[t][b][k] = exp(tanh( sum_h d[b][h] * mem[t][k][h] ))
// Batched 32x32 WMMA GEMM over t. Grid: T_ blocks x 128 threads (4 waves).
// ---------------------------------------------------------------------------
__global__ __launch_bounds__(128) void beta_kernel(
    const bfbits* __restrict__ dbf,     // [B,H]
    const bfbits* __restrict__ membf,   // [T,B,H]
    float* __restrict__ beta) {         // [T,B,B]
  const int t    = blockIdx.x;
  const int w    = threadIdx.x >> 5;
  const int lane = threadIdx.x & 31;
  const int mtile = w & 1;
  const int ntile = w >> 1;

  const bfbits* Arow = dbf + (size_t)mtile * 16 * H_;
  const bfbits* Bt   = membf + ((size_t)t * B_ + ntile * 16) * H_;
  v8f acc = zero8();
#pragma unroll 1
  for (int kk = 0; kk < H_; kk += 32) {
    v16bf a = load_a_tile(Arow + kk, H_, lane);
    v16bf b = load_b_tile(Bt + kk, H_, lane);
    acc = wmma_bf16(a, b, acc);
  }
  const int n  = ntile * 16 + (lane & 15);
  const int mb = mtile * 16 + (lane >> 4) * 8;
#pragma unroll
  for (int r = 0; r < 8; ++r) {
    int m = mb + r;
    beta[((size_t)t * B_ + m) * B_ + n] = __expf(tanhf(acc[r]));
  }
}

// Normalize beta over t, write transposed bf16: betaT[b][t*32+k].
__global__ __launch_bounds__(128) void beta_norm_kernel(
    const float* __restrict__ beta, bfbits* __restrict__ betaT) {
  const int b = blockIdx.x >> 5;
  const int k = blockIdx.x & 31;
  __shared__ float red[128];
  float s = 0.0f;
  for (int t = threadIdx.x; t < T_; t += 128)
    s += beta[((size_t)t * B_ + b) * B_ + k];
  red[threadIdx.x] = s;
  __syncthreads();
  for (int st = 64; st > 0; st >>= 1) {
    if (threadIdx.x < st) red[threadIdx.x] += red[threadIdx.x + st];
    __syncthreads();
  }
  const float inv = 1.0f / red[0];
  for (int t = threadIdx.x; t < T_; t += 128)
    betaT[(size_t)b * TK + (size_t)t * B_ + k] =
        f2bf(beta[((size_t)t * B_ + b) * B_ + k] * inv);
}

// mem = memF + memB; emit fp32 + bf16 [t][b][h] + transposed bf16 [h][t*32+b]
__global__ __launch_bounds__(256) void mem_post_kernel(
    const float* __restrict__ memF, const float* __restrict__ memB,
    float* __restrict__ mem, bfbits* __restrict__ mem_bf,
    bfbits* __restrict__ memT) {
  size_t i = (size_t)blockIdx.x * 256 + threadIdx.x;
  if (i >= (size_t)T_ * B_ * H_) return;
  int h = (int)(i & (H_ - 1));
  size_t tb = i >> 9;
  int b = (int)(tb & (B_ - 1));
  int t = (int)(tb >> 5);
  float v = memF[i] + memB[i];
  bfbits vb = f2bf(v);
  mem[i] = v;
  mem_bf[i] = vb;
  memT[(size_t)h * TK + (size_t)t * B_ + b] = vb;
}

// h0 = hT_f + hT_b ; c0 = 0
__global__ __launch_bounds__(256) void h0_kernel(
    const float* __restrict__ hf, const float* __restrict__ hb,
    float* __restrict__ h0, bfbits* __restrict__ h0bf, float* __restrict__ c0) {
  int i = blockIdx.x * 256 + threadIdx.x;
  if (i >= B_ * H_) return;
  float v = hf[i] + hb[i];
  h0[i] = v;
  h0bf[i] = f2bf(v);
  c0[i] = 0.0f;
}

// log_softmax over V + argmax. Grid: B_ blocks x 256 threads.
__global__ __launch_bounds__(256) void lsm_argmax_kernel(
    const float* __restrict__ logits, float* __restrict__ props, int s,
    int* __restrict__ wtok) {
  const int b = blockIdx.x;
  const float* row = logits + (size_t)b * V_;
  __shared__ float smax[256];
  __shared__ int sidx[256];
  float mx = -3.4e38f;
  int mi = 0;
  for (int v = threadIdx.x; v < V_; v += 256) {
    float x = row[v];
    if (x > mx) { mx = x; mi = v; }
  }
  smax[threadIdx.x] = mx;
  sidx[threadIdx.x] = mi;
  __syncthreads();
  for (int st = 128; st > 0; st >>= 1) {
    if (threadIdx.x < st) {
      float o = smax[threadIdx.x + st];
      int oi = sidx[threadIdx.x + st];
      if (o > smax[threadIdx.x] ||
          (o == smax[threadIdx.x] && oi < sidx[threadIdx.x])) {
        smax[threadIdx.x] = o;
        sidx[threadIdx.x] = oi;
      }
    }
    __syncthreads();
  }
  const float gmax = smax[0];
  const int gidx = sidx[0];
  __syncthreads();
  float se = 0.0f;
  for (int v = threadIdx.x; v < V_; v += 256) se += __expf(row[v] - gmax);
  smax[threadIdx.x] = se;
  __syncthreads();
  for (int st = 128; st > 0; st >>= 1) {
    if (threadIdx.x < st) smax[threadIdx.x] += smax[threadIdx.x + st];
    __syncthreads();
  }
  const float lse = __logf(smax[0]);
  float* out = props + ((size_t)b * S_ + s) * V_;
  for (int v = threadIdx.x; v < V_; v += 256) out[v] = row[v] - gmax - lse;
  if (threadIdx.x == 0) wtok[b] = gidx;
}

// Classifier: out[b][l] = sum_j r[b][j]*cls_W[l][j] + cls_b[l]
__global__ __launch_bounds__(256) void classifier_kernel(
    const float* __restrict__ vts, const float* __restrict__ mem,
    const float* __restrict__ clsW, const float* __restrict__ clsb,
    float* __restrict__ out) {
  const int b = blockIdx.x / L_;
  const int l = blockIdx.x % L_;
  const int KC = (T_ + S_) * H_;     // 237568
  const float* wrow = clsW + (size_t)l * KC;
  float s = 0.0f;
  for (int j = threadIdx.x; j < KC; j += 256) {
    float rv;
    if (j < S_ * H_) {
      int si = j >> 9, h = j & (H_ - 1);
      rv = vts[((size_t)si * B_ + b) * H_ + h];
    } else {
      int j2 = j - S_ * H_;
      int t = j2 >> 9, h = j2 & (H_ - 1);
      rv = mem[((size_t)t * B_ + b) * H_ + h];
    }
    s += rv * wrow[j];
  }
  __shared__ float red[256];
  red[threadIdx.x] = s;
  __syncthreads();
  for (int st = 128; st > 0; st >>= 1) {
    if (threadIdx.x < st) red[threadIdx.x] += red[threadIdx.x + st];
    __syncthreads();
  }
  if (threadIdx.x == 0) out[(size_t)b * L_ + l] = red[0] + clsb[l];
}

// ---------------------------------------------------------------------------
// Host orchestration
// ---------------------------------------------------------------------------
static inline int blocks_for(size_t n) { return (int)((n + 255) / 256); }

extern "C" void kernel_launch(void* const* d_in, const int* in_sizes, int n_in,
                              void* d_out, int out_size, void* d_ws, size_t ws_size,
                              hipStream_t stream) {
  (void)in_sizes; (void)n_in; (void)out_size; (void)ws_size;

  const int*   original = (const int*)  d_in[0];
  const float* emb_w    = (const float*)d_in[1];
  const float* Wih_f    = (const float*)d_in[2];
  const float* Whh_f    = (const float*)d_in[3];
  const float* b_f      = (const float*)d_in[4];
  const float* Wih_b    = (const float*)d_in[5];
  const float* Whh_b    = (const float*)d_in[6];
  const float* b_b      = (const float*)d_in[7];
  const float* dWih     = (const float*)d_in[8];
  const float* dWhh     = (const float*)d_in[9];
  const float* d_bias   = (const float*)d_in[10];
  const float* sAtt     = (const float*)d_in[11];
  const float* cAtt     = (const float*)d_in[12];
  const float* genW     = (const float*)d_in[13];
  const float* genb     = (const float*)d_in[14];
  const float* clsW     = (const float*)d_in[15];
  const float* clsb     = (const float*)d_in[16];

  float* out_props = (float*)d_out;                       // [B,S,V]
  float* out_cls   = out_props + (size_t)B_ * S_ * V_;    // [B,L]

  // ---- workspace bump allocator -------------------------------------------
  char* ws = (char*)d_ws;
  size_t off = 0;
  auto alloc = [&](size_t bytes) -> void* {
    off = (off + 255) & ~(size_t)255;
    void* p = ws + off;
    off += bytes;
    return p;
  };
  const size_t WSZ = (size_t)G4H * H_;      // 2048*512
  const size_t BH  = (size_t)B_ * H_;       // 32*512
  const size_t TBH = (size_t)T_ * B_ * H_;  // 400*32*512

  bfbits* WihF_bf = (bfbits*)alloc(WSZ * 2);
  bfbits* WhhF_bf = (bfbits*)alloc(WSZ * 2);
  bfbits* WihB_bf = (bfbits*)alloc(WSZ * 2);
  bfbits* WhhB_bf = (bfbits*)alloc(WSZ * 2);
  bfbits* dWih_bf = (bfbits*)alloc(WSZ * 2);
  bfbits* dWhh_bf = (bfbits*)alloc(WSZ * 2);
  bfbits* sAtt_bf = (bfbits*)alloc((size_t)H_ * H_ * 2);
  bfbits* cAtt_bf = (bfbits*)alloc((size_t)H_ * H_ * 2);
  bfbits* genW_bf = (bfbits*)alloc((size_t)V_ * H_ * 2);
  bfbits* xs_bf   = (bfbits*)alloc(TBH * 2);
  float*  memF    = (float*) alloc(TBH * 4);
  float*  memB    = (float*) alloc(TBH * 4);
  float*  mem_f   = (float*) alloc(TBH * 4);
  bfbits* mem_bf  = (bfbits*)alloc(TBH * 2);
  bfbits* memT_bf = (bfbits*)alloc(TBH * 2);
  float*  hTf     = (float*) alloc(BH * 4);
  float*  hTb     = (float*) alloc(BH * 4);
  float*  hD      = (float*) alloc(BH * 4);
  bfbits* hD_bf   = (bfbits*)alloc(BH * 2);
  float*  cD      = (float*) alloc(BH * 4);
  bfbits* xD_bf   = (bfbits*)alloc(BH * 2);
  bfbits* dvec_bf = (bfbits*)alloc(BH * 2);
  float*  beta_f  = (float*) alloc((size_t)T_ * B_ * B_ * 4);
  bfbits* betaT   = (bfbits*)alloc((size_t)B_ * TK * 2);
  float*  vC      = (float*) alloc(BH * 4);
  bfbits* vC_bf   = (bfbits*)alloc(BH * 2);
  float*  vts     = (float*) alloc((size_t)S_ * B_ * H_ * 4);
  float*  logits  = (float*) alloc((size_t)B_ * V_ * 4);
  int*    wtok    = (int*)   alloc(B_ * sizeof(int));

  // ---- one-time conversions ----------------------------------------------
  f2bf_kernel<<<blocks_for(WSZ), 256, 0, stream>>>(Wih_f, WihF_bf, WSZ);
  f2bf_kernel<<<blocks_for(WSZ), 256, 0, stream>>>(Whh_f, WhhF_bf, WSZ);
  f2bf_kernel<<<blocks_for(WSZ), 256, 0, stream>>>(Wih_b, WihB_bf, WSZ);
  f2bf_kernel<<<blocks_for(WSZ), 256, 0, stream>>>(Whh_b, WhhB_bf, WSZ);
  f2bf_kernel<<<blocks_for(WSZ), 256, 0, stream>>>(dWih, dWih_bf, WSZ);
  f2bf_kernel<<<blocks_for(WSZ), 256, 0, stream>>>(dWhh, dWhh_bf, WSZ);
  f2bf_kernel<<<blocks_for((size_t)H_ * H_), 256, 0, stream>>>(sAtt, sAtt_bf, (size_t)H_ * H_);
  f2bf_kernel<<<blocks_for((size_t)H_ * H_), 256, 0, stream>>>(cAtt, cAtt_bf, (size_t)H_ * H_);
  f2bf_kernel<<<blocks_for((size_t)V_ * H_), 256, 0, stream>>>(genW, genW_bf, (size_t)V_ * H_);
  embed_all_kernel<<<blocks_for(TBH), 256, 0, stream>>>(original, emb_w, xs_bf);

  // ---- encoder: both directions persistently, concurrently ---------------
  lstm_seq_kernel<<<2, 1024, 0, stream>>>(
      xs_bf, WihF_bf, WhhF_bf, b_f, WihB_bf, WhhB_bf, b_b,
      memF, memB, hTf, hTb);
  mem_post_kernel<<<blocks_for(TBH), 256, 0, stream>>>(memF, memB,
                                                       mem_f, mem_bf, memT_bf);
  h0_kernel<<<blocks_for(BH), 256, 0, stream>>>(hTf, hTb, hD, hD_bf, cD);

  // ---- decoder ------------------------------------------------------------
  const int g512 = ((H_ / 16) * 2) / 8;   // 8 blocks for N=512
  const int gGen = ((V_ / 16) * 2) / 8;   // 500 blocks for N=32000
  for (int s = 0; s < S_; ++s) {
    embed_step_kernel<<<blocks_for(BH), 256, 0, stream>>>(wtok, emb_w, xD_bf, s == 0);
    lstm_cell_kernel<<<4, 256, 0, stream>>>(
        xD_bf, hD_bf, dWih_bf, dWhh_bf, d_bias, cD, hD, hD_bf);

    // mva(summ_att) -> v_c
    gemm_m32_kernel<<<g512, 256, 0, stream>>>(hD_bf, sAtt_bf, nullptr,
                                              nullptr, dvec_bf, H_, H_);
    beta_kernel<<<T_, 128, 0, stream>>>(dvec_bf, mem_bf, beta_f);
    beta_norm_kernel<<<B_ * B_, 128, 0, stream>>>(beta_f, betaT);
    gemm_m32_kernel<<<g512, 256, 0, stream>>>(betaT, memT_bf, nullptr,
                                              vC, vC_bf, H_, TK);

    // mva(cls_att) -> v_t (stored for classifier)
    gemm_m32_kernel<<<g512, 256, 0, stream>>>(hD_bf, cAtt_bf, nullptr,
                                              nullptr, dvec_bf, H_, H_);
    beta_kernel<<<T_, 128, 0, stream>>>(dvec_bf, mem_bf, beta_f);
    beta_norm_kernel<<<B_ * B_, 128, 0, stream>>>(beta_f, betaT);
    gemm_m32_kernel<<<g512, 256, 0, stream>>>(betaT, memT_bf, nullptr,
                                              vts + (size_t)s * BH, nullptr, H_, TK);

    // generator + log_softmax + argmax feedback
    gemm_m32_kernel<<<gGen, 256, 0, stream>>>(vC_bf, genW_bf, genb,
                                              logits, nullptr, V_, H_);
    lsm_argmax_kernel<<<B_, 256, 0, stream>>>(logits, out_props, s, wtok);
  }

  // ---- classifier ---------------------------------------------------------
  classifier_kernel<<<B_ * L_, 256, 0, stream>>>(vts, mem_f, clsW, clsb, out_cls);
}
